// qKv_attention_12086037971050
// MI455X (gfx1250) — matrix operand order repaired and checked by reading, not compiled, not run on hardware
//
#include <hip/hip_runtime.h>
#include <hip/hip_bf16.h>
#include <cstdint>

// RBF-kernel attention for MI455X (gfx1250), bf16 WMMA pipeline with
// async-to-LDS panel staging (ASYNCcnt) feeding the matrix pipes.
// Stages: convert -> QKV GEMM -> norms -> attention -> projection GEMM.

typedef __bf16 bf16_t;
typedef __attribute__((ext_vector_type(16))) __bf16 v16bf;
typedef __attribute__((ext_vector_type(8)))  __bf16 v8bf;
typedef __attribute__((ext_vector_type(8)))  float  v8f;
typedef __attribute__((ext_vector_type(4)))  int    v4i;

#define B_    2
#define T_    2048
#define C_    1024
#define DIM_  1024
#define H_    16
#define HD_   64
#define BT_   (B_ * T_)      // 4096 rows
#define NQKV_ (3 * DIM_)     // 3072

#define WMMA_BF16(a, b, c) \
  __builtin_amdgcn_wmma_f32_16x16x32_bf16(false, (a), false, (b), (short)0, (c), false, false)

// Async copy of 16 bytes global -> LDS (GLOBAL_LOAD_ASYNC_TO_LDS_B128).
// Toolchain prototype takes generic v4i* pointers; order follows the
// load_*_lds builtin family: (global src, LDS dst, imm offset, imm cpol).
#define ASYNC_CP16(dst, src)                                   \
  __builtin_amdgcn_global_load_async_to_lds_b128(              \
      (v4i*)(src), (v4i*)(dst), 0, 0)

#define WAIT_ASYNC() asm volatile("s_wait_asynccnt 0" ::: "memory")

static __device__ __forceinline__ v8f zerov8() {
  v8f z = {0.f, 0.f, 0.f, 0.f, 0.f, 0.f, 0.f, 0.f};
  return z;
}

// A fragment: 16x32 bf16 (MxK). lane m = lane&15; per-lane k values are the
// runs [half*8 .. +7] and [16+half*8 .. +7] (ISA 16-bit A layout, wave32).
static __device__ __forceinline__ v16bf load_a_frag(const bf16_t* arow, int k0, int half) {
  union { v16bf v; v8bf h[2]; } u;
  u.h[0] = *reinterpret_cast<const v8bf*>(arow + k0 + half * 8);
  u.h[1] = *reinterpret_cast<const v8bf*>(arow + k0 + 16 + half * 8);
  return u.v;
}

// B fragment: 32x16 bf16 (KxN). Per-lane column n fixed (folded into brow by
// caller), per-lane k values are the runs [half*8 .. +7] and [16+half*8 .. +7].
static __device__ __forceinline__ v16bf load_b_frag(const bf16_t* brow, int k0, int half) {
  union { v16bf v; v8bf h[2]; } u;
  u.h[0] = *reinterpret_cast<const v8bf*>(brow + k0 + half * 8);
  u.h[1] = *reinterpret_cast<const v8bf*>(brow + k0 + 16 + half * 8);
  return u.v;
}

// Block GEMM: 4 waves, wave computes rows [m0, m0+16) (caller adds 16*wave to m0) x cols [n0, n0+64).
// The 64x32 B^T panel chunk for each k-step is staged once per block into LDS
// with async copies and shared by all 4 waves.
static __device__ __forceinline__ void block_gemm_16x64(
    const bf16_t* __restrict__ A, int lda, int m0,
    const bf16_t* __restrict__ BT, int ldb, int n0,
    int K, bf16_t (*bs)[32], v8f acc[4]) {
  const int tid  = threadIdx.x;
  const int lane = tid & 31;
  const int half = lane >> 4;
  const int r16  = lane & 15;
  const bf16_t* arow = A + (size_t)(m0 + r16) * lda;

  const int brow = tid >> 1;          // 0..63: B^T row (n)
  const int bpart = tid & 1;          // half-row of 16 elements
  const bf16_t* bsrc = BT + (size_t)(n0 + brow) * ldb + bpart * 16;
  bf16_t* bdst = &bs[brow][bpart * 16];

  for (int k0 = 0; k0 < K; k0 += 32) {
    __syncthreads();                  // all waves done reading previous chunk
    ASYNC_CP16(bdst,     bsrc + k0);
    ASYNC_CP16(bdst + 8, bsrc + k0 + 8);
    if (k0 + 32 < K) {
      __builtin_prefetch(arow + k0 + 32, 0, 1);   // global_prefetch_b8
    }
    v16bf a = load_a_frag(arow, k0, half);
    WAIT_ASYNC();
    __syncthreads();                  // panel visible to all waves
#pragma unroll
    for (int sub = 0; sub < 4; ++sub) {
      v16bf b = load_b_frag(&bs[16 * sub + r16][0], 0, half);
      acc[sub] = WMMA_BF16(a, b, acc[sub]);
    }
  }
}

// ---------------- Stage 0: conversions ----------------

__global__ void cvt_f32_bf16_kernel(const float* __restrict__ src,
                                    bf16_t* __restrict__ dst, int n) {
  int i = blockIdx.x * blockDim.x + threadIdx.x;
  if (i < n) dst[i] = (bf16_t)src[i];
}

// src is K x N row-major f32; dst is N x K row-major bf16 (i.e. B^T).
__global__ void cvt_transpose_kernel(const float* __restrict__ src,
                                     bf16_t* __restrict__ dst, int K, int N) {
  int i = blockIdx.x * blockDim.x + threadIdx.x;
  if (i < K * N) {
    int k = i / N;
    int n = i - k * N;
    dst[(size_t)n * K + k] = (bf16_t)src[i];
  }
}

// ---------------- Stage 1: QKV GEMM ----------------
// qkv = x(4096x1024) @ Wqkv(1024x3072) + bqkv; epilogue splits heads:
//   q,k -> [b][h][t][d] row-major (t x 64);  v -> vT [b][h][d][t] (64 x 2048)

__global__ __launch_bounds__(128) void qkv_gemm_kernel(
    const bf16_t* __restrict__ xb, const bf16_t* __restrict__ WqkvT,
    const float* __restrict__ bqkv,
    bf16_t* __restrict__ qbuf, bf16_t* __restrict__ kbuf,
    bf16_t* __restrict__ vT) {
  __shared__ __align__(16) bf16_t bs[64][32];

  const int wave = threadIdx.x >> 5;
  const int lane = threadIdx.x & 31;
  const int half = lane >> 4;
  const int r16  = lane & 15;
  const int n0 = blockIdx.x * 64;
  const int m0 = blockIdx.y * 64 + wave * 16;

  v8f acc[4] = {zerov8(), zerov8(), zerov8(), zerov8()};
  block_gemm_16x64(xb, C_, m0, WqkvT, C_, n0, C_, bs, acc);

#pragma unroll
  for (int sub = 0; sub < 4; ++sub) {
    const int n = n0 + 16 * sub + r16;
    const float bias = bqkv[n];
    const int part = n >> 10;   // 0:q 1:k 2:v
    const int col  = n & 1023;
    const int h = col >> 6;
    const int d = col & 63;
#pragma unroll
    for (int r = 0; r < 8; ++r) {
      const int m = m0 + r + 8 * half;   // global row (C layout: M = r + 8*half)
      const int b = m >> 11;
      const int t = m & 2047;
      const bf16_t val = (bf16_t)(acc[sub][r] + bias);
      const size_t head = (size_t)(b * H_ + h);
      if (part == 0) {
        qbuf[(head * T_ + t) * HD_ + d] = val;
      } else if (part == 1) {
        kbuf[(head * T_ + t) * HD_ + d] = val;
      } else {
        vT[(head * HD_ + d) * T_ + t] = val;
      }
    }
  }
}

// ---------------- Stage 2a: row norms ----------------

__global__ void norms_kernel(const bf16_t* __restrict__ qbuf,
                             const bf16_t* __restrict__ kbuf,
                             float* __restrict__ qn, float* __restrict__ kn) {
  int tid = blockIdx.x * blockDim.x + threadIdx.x;  // 0 .. B*H*T-1
  const bf16_t* qr = qbuf + (size_t)tid * HD_;
  const bf16_t* kr = kbuf + (size_t)tid * HD_;
  float sq = 0.f, sk = 0.f;
#pragma unroll
  for (int i = 0; i < HD_ / 8; ++i) {
    v8bf a = *reinterpret_cast<const v8bf*>(qr + i * 8);
    v8bf b = *reinterpret_cast<const v8bf*>(kr + i * 8);
#pragma unroll
    for (int e = 0; e < 8; ++e) {
      float fa = (float)a[e], fb = (float)b[e];
      sq += fa * fa;
      sk += fb * fb;
    }
  }
  qn[tid] = sq;
  kn[tid] = sk;
}

// ---------------- Stage 2b: RBF attention ----------------
// out[i,:] = sum_j exp(-(kn_i + qn_j - 2 k_i.q_j) * sigma) * v[j,:]
// grid: x = b*H (32), y = i-tile of 64 rows (32); 4 waves, 16 rows per wave.
// Per j-chunk of 32, the q (32x64) and v (64x32) panels are shared by all 4
// waves -> staged once into LDS via async copies.

__global__ __launch_bounds__(128) void attn_kernel(
    const bf16_t* __restrict__ qbuf, const bf16_t* __restrict__ kbuf,
    const bf16_t* __restrict__ vT, const float* __restrict__ qn,
    const float* __restrict__ kn, const float* __restrict__ rsig,
    bf16_t* __restrict__ attn) {
  __shared__ __align__(16) bf16_t q_s[32][64];      // rows j0..j0+31
  __shared__ __align__(16) bf16_t v_s[64][32];      // rows d=0..63, cols j
  __shared__ __align__(16) bf16_t slds[4][16][32];  // per-wave score tile

  const int tid  = threadIdx.x;
  const int wave = tid >> 5;
  const int lane = tid & 31;
  const int half = lane >> 4;
  const int r16  = lane & 15;
  const int bh = blockIdx.x;
  const int b  = bh >> 4;
  const int h  = bh & 15;
  const int i0 = blockIdx.y * 64 + wave * 16;
  const float sigma = *rsig;

  const bf16_t* kh  = kbuf + (size_t)bh * T_ * HD_;
  const bf16_t* qh  = qbuf + (size_t)bh * T_ * HD_;
  const bf16_t* vh  = vT   + (size_t)bh * HD_ * T_;
  const float*  qnh = qn + (size_t)bh * T_;
  const float*  knh = kn + (size_t)bh * T_;

  // A fragments: k rows of this i-tile, d = 0..31 and 32..63 (resident whole loop)
  const bf16_t* krow = kh + (size_t)(i0 + r16) * HD_;
  v16bf ak0 = load_a_frag(krow, 0, half);
  v16bf ak1 = load_a_frag(krow, 32, half);

  float knv[8];
#pragma unroll
  for (int r = 0; r < 8; ++r) knv[r] = knh[i0 + r + 8 * half];

  // async-copy index precompute
  bf16_t* qdst = &q_s[0][0] + tid * 16;               // q chunk is contiguous
  const int vrow_cp = tid >> 1, vpart = tid & 1;
  const bf16_t* vsrc0 = vh + (size_t)vrow_cp * T_ + vpart * 16;
  bf16_t* vdst = &v_s[vrow_cp][vpart * 16];

  v8f acc[4] = {zerov8(), zerov8(), zerov8(), zerov8()};

  for (int j0 = 0; j0 < T_; j0 += 32) {
    __syncthreads();  // all waves done with previous chunk
    {
      const bf16_t* qsrc = qh + (size_t)j0 * HD_ + tid * 16;
      ASYNC_CP16(qdst,     qsrc);
      ASYNC_CP16(qdst + 8, qsrc + 8);
      const bf16_t* vsrc = vsrc0 + j0;
      ASYNC_CP16(vdst,     vsrc);
      ASYNC_CP16(vdst + 8, vsrc + 8);
    }
    WAIT_ASYNC();
    __syncthreads();  // panels visible to all waves

    // ---- scores: two 16x16 tiles over columns j0..j0+31 ----
#pragma unroll
    for (int jt = 0; jt < 2; ++jt) {
      const bf16_t* qrow = &q_s[16 * jt + r16][0];  // B^T row = q row (contig d)
      v16bf bq0 = load_b_frag(qrow, 0, half);
      v16bf bq1 = load_b_frag(qrow, 32, half);
      v8f s = zerov8();
      s = WMMA_BF16(ak0, bq0, s);
      s = WMMA_BF16(ak1, bq1, s);
      const float qnv = qnh[j0 + 16 * jt + r16];
#pragma unroll
      for (int r = 0; r < 8; ++r) {
        float d2 = knv[r] + qnv - 2.0f * s[r];
        float w  = __expf(-d2 * sigma);
        slds[wave][r + 8 * half][16 * jt + r16] = (bf16_t)w;  // (m, n) C layout
      }
    }
    // per-wave LDS RAW: DS ops are in-order, but be explicit.
    asm volatile("s_wait_dscnt 0" ::: "memory");

    // ---- reload scores as an A fragment (16x32, K = j) ----
    union { v16bf v; v8bf h2[2]; } ua;
    const bf16_t* srow = &slds[wave][r16][0];
    ua.h2[0] = *reinterpret_cast<const v8bf*>(srow + half * 8);
    ua.h2[1] = *reinterpret_cast<const v8bf*>(srow + 16 + half * 8);

    // ---- accumulate out += S(16x32) @ V(32x64) ----
#pragma unroll
    for (int sub = 0; sub < 4; ++sub) {
      v16bf bv = load_b_frag(&v_s[16 * sub + r16][0], 0, half);
      acc[sub] = WMMA_BF16(ua.v, bv, acc[sub]);
    }
  }

  // epilogue: attn[b][t][h*64 + d], bf16 row-major (4096 x 1024)
#pragma unroll
  for (int sub = 0; sub < 4; ++sub) {
#pragma unroll
    for (int r = 0; r < 8; ++r) {
      const int t = i0 + r + 8 * half;
      const int d = 16 * sub + r16;
      attn[((size_t)(b * T_ + t)) * DIM_ + h * HD_ + d] = (bf16_t)acc[sub][r];
    }
  }
}

// ---------------- Stage 3: projection GEMM ----------------

__global__ __launch_bounds__(128) void proj_gemm_kernel(
    const bf16_t* __restrict__ attn, const bf16_t* __restrict__ WprojT,
    const float* __restrict__ bproj, float* __restrict__ out) {
  __shared__ __align__(16) bf16_t bs[64][32];

  const int wave = threadIdx.x >> 5;
  const int lane = threadIdx.x & 31;
  const int half = lane >> 4;
  const int r16  = lane & 15;
  const int n0 = blockIdx.x * 64;
  const int m0 = blockIdx.y * 64 + wave * 16;

  v8f acc[4] = {zerov8(), zerov8(), zerov8(), zerov8()};
  block_gemm_16x64(attn, DIM_, m0, WprojT, DIM_, n0, DIM_, bs, acc);

#pragma unroll
  for (int sub = 0; sub < 4; ++sub) {
    const int n = n0 + 16 * sub + r16;
    const float bias = bproj[n];
#pragma unroll
    for (int r = 0; r < 8; ++r) {
      const int m = m0 + r + 8 * half;
      out[(size_t)m * C_ + n] = acc[sub][r] + bias;
    }
  }
}

// ---------------- host-side launch ----------------

extern "C" void kernel_launch(void* const* d_in, const int* in_sizes, int n_in,
                              void* d_out, int out_size, void* d_ws, size_t ws_size,
                              hipStream_t stream) {
  const float* x     = (const float*)d_in[0];
  const float* Wqkv  = (const float*)d_in[1];
  const float* bqkv  = (const float*)d_in[2];
  const float* rsig  = (const float*)d_in[3];
  const float* Wproj = (const float*)d_in[4];
  const float* bproj = (const float*)d_in[5];
  float* out = (float*)d_out;

  size_t off = 0;
  auto wsalloc = [&](size_t bytes) -> void* {
    void* p = (char*)d_ws + off;
    off = (off + bytes + 255) & ~(size_t)255;
    return p;
  };

  bf16_t* xb     = (bf16_t*)wsalloc((size_t)BT_ * C_ * sizeof(bf16_t));     // 8 MB
  bf16_t* WqkvT  = (bf16_t*)wsalloc((size_t)NQKV_ * C_ * sizeof(bf16_t));   // 6 MB
  bf16_t* WprojT = (bf16_t*)wsalloc((size_t)C_ * DIM_ * sizeof(bf16_t));    // 2 MB
  bf16_t* qbuf   = (bf16_t*)wsalloc((size_t)B_ * H_ * T_ * HD_ * sizeof(bf16_t));
  bf16_t* kbuf   = (bf16_t*)wsalloc((size_t)B_ * H_ * T_ * HD_ * sizeof(bf16_t));
  bf16_t* vT     = (bf16_t*)wsalloc((size_t)B_ * H_ * HD_ * T_ * sizeof(bf16_t));
  float*  qn     = (float*)wsalloc((size_t)B_ * H_ * T_ * sizeof(float));
  float*  kn     = (float*)wsalloc((size_t)B_ * H_ * T_ * sizeof(float));
  bf16_t* attn   = (bf16_t*)wsalloc((size_t)BT_ * DIM_ * sizeof(bf16_t));   // 8 MB
  (void)ws_size;

  // Stage 0: conversions
  {
    int n = BT_ * C_;
    cvt_f32_bf16_kernel<<<(n + 255) / 256, 256, 0, stream>>>(x, xb, n);
  }
  {
    int n = C_ * NQKV_;
    cvt_transpose_kernel<<<(n + 255) / 256, 256, 0, stream>>>(Wqkv, WqkvT, C_, NQKV_);
  }
  {
    int n = DIM_ * C_;
    cvt_transpose_kernel<<<(n + 255) / 256, 256, 0, stream>>>(Wproj, WprojT, DIM_, C_);
  }

  // Stage 1: QKV GEMM (M=4096, N=3072, K=1024)
  qkv_gemm_kernel<<<dim3(NQKV_ / 64, BT_ / 64), 128, 0, stream>>>(
      xb, WqkvT, bqkv, qbuf, kbuf, vT);

  // Stage 2a: norms over head_dim
  norms_kernel<<<(B_ * H_ * T_) / 256, 256, 0, stream>>>(qbuf, kbuf, qn, kn);

  // Stage 2b: attention (grid: heads x i-tiles)
  attn_kernel<<<dim3(B_ * H_, T_ / 64), 128, 0, stream>>>(
      qbuf, kbuf, vT, qn, kn, rsig, attn);

  // Stage 3: projection GEMM (M=4096, N=1024, K=1024)
  proj_gemm_kernel<<<dim3(C_ / 64, BT_ / 64), 128, 0, stream>>>(
      attn, WprojT, bproj, out);
}
